// BatchLinearMasked_47785806135497
// MI455X (gfx1250) — matrix operand order repaired and checked by reading, not compiled, not run on hardware
//
#include <hip/hip_runtime.h>
#include <hip/hip_bf16.h>
#include <string.h>

typedef __attribute__((ext_vector_type(16))) __bf16 v16bf;
typedef __attribute__((ext_vector_type(8)))  __bf16 v8bf;
typedef __attribute__((ext_vector_type(8)))  float  v8f;
typedef __attribute__((ext_vector_type(4)))  unsigned int u32x4;
typedef __attribute__((ext_vector_type(4)))  int v4i;
typedef unsigned short ushort_t;

#define AS1 __attribute__((address_space(1)))
#define AS3 __attribute__((address_space(3)))

#define B_TOT 2048
#define N_GRP 64
#define D_DIM 256
#define BT 32            // batch rows per workgroup
#define WPAD 264         // padded LDS row stride (elements): 528B -> 4-bank lane stride
#define TPB 256

#if __has_builtin(__builtin_amdgcn_global_load_async_to_lds_b128) && \
    __has_builtin(__builtin_amdgcn_s_wait_asynccnt)
#define HAVE_ASYNC_LDS 1
#else
#define HAVE_ASYNC_LDS 0
#endif

static __device__ __forceinline__ v8f wmma_bf16(v16bf a, v16bf b, v8f c) {
  return __builtin_amdgcn_wmma_f32_16x16x32_bf16(false, a, false, b, (short)0, c,
                                                 false, false);
}

// split fp32 into bf16 hi (truncated, exact) + bf16 lo (RNE of remainder)
static __device__ __forceinline__ void split_bits(float v, ushort_t& hb, ushort_t& lb) {
  unsigned u = __float_as_uint(v);
  hb = (ushort_t)(u >> 16);
  float fh = __uint_as_float(u & 0xFFFF0000u);
  __bf16 l = (__bf16)(v - fh);
  lb = __builtin_bit_cast(ushort_t, l);
}

// ---------------- prep: masked W -> bf16 hi/lo pair in workspace ----------------
__global__ void prep_w_kernel(const float* __restrict__ w, const float* __restrict__ wm,
                              ushort_t* __restrict__ wh, ushort_t* __restrict__ wl,
                              int n) {
  int i = blockIdx.x * blockDim.x + threadIdx.x;
  if (i < n) {
    float v = w[i] * wm[i];
    ushort_t hb, lb;
    split_bits(v, hb, lb);
    wh[i] = hb;
    wl[i] = lb;
  }
}

__global__ void prep_b_kernel(const float* __restrict__ b, const float* __restrict__ bm,
                              float* __restrict__ bout, int n) {
  int i = blockIdx.x * blockDim.x + threadIdx.x;
  if (i < n) bout[i] = b[i] * bm[i];
}

// ---------------- fused 4-step recurrence, split-bf16 WMMA ----------------
// 8 waves; each wave: MT=2 m-tiles (rows 0..31) x JT=2 j-tiles (32-col range)
__global__ __launch_bounds__(TPB)
void BatchLinearMasked_recur_kernel(const float* __restrict__ x,
                                    const ushort_t* __restrict__ gWh,
                                    const ushort_t* __restrict__ gWl,
                                    const float* __restrict__ gB,
                                    float* __restrict__ out) {
  // 256*264*2*2 + 32*264*2*2 = 270336 + 33792 = 304128 B (< 320 KB WGP LDS)
  __shared__ __align__(16) ushort_t sWh[D_DIM * WPAD];
  __shared__ __align__(16) ushort_t sWl[D_DIM * WPAD];
  __shared__ __align__(16) ushort_t sXh[BT * WPAD];
  __shared__ __align__(16) ushort_t sXl[BT * WPAD];

  const int n    = blockIdx.y;
  const int b0   = blockIdx.x * BT;
  const int t    = threadIdx.x;
  const int lane = t & 31;
  const int jr   = t >> 5;        // wave 0..7 -> j-range of 32 cols
  const int ln15 = lane & 15;
  const int lhi  = lane >> 4;     // 0 or 1

  // ---- stage masked split-W into LDS (padded rows) ----
  const ushort_t* wsrcH = gWh + (size_t)n * (D_DIM * D_DIM);
  const ushort_t* wsrcL = gWl + (size_t)n * (D_DIM * D_DIM);
#if HAVE_ASYNC_LDS
  // async DMA: no VGPR round trip, overlaps with x staging + cur loads below
#pragma unroll 4
  for (int k = 0; k < 32; ++k) {
    int c   = k * TPB + t;
    int row = c >> 5;
    int cw  = (c & 31) * 8;
    __builtin_amdgcn_global_load_async_to_lds_b128(
        (AS1 v4i*)&wsrcH[row * D_DIM + cw],
        (AS3 v4i*)&sWh[row * WPAD + cw], 0, 0);
    __builtin_amdgcn_global_load_async_to_lds_b128(
        (AS1 v4i*)&wsrcL[row * D_DIM + cw],
        (AS3 v4i*)&sWl[row * WPAD + cw], 0, 0);
  }
#else
#pragma unroll 4
  for (int k = 0; k < 32; ++k) {
    int c   = k * TPB + t;        // chunk id: 256 rows x 32 chunks of 8 elems
    int row = c >> 5;
    int cw  = (c & 31) * 8;
    *(u32x4*)&sWh[row * WPAD + cw] = *(const u32x4*)&wsrcH[row * D_DIM + cw];
    *(u32x4*)&sWl[row * WPAD + cw] = *(const u32x4*)&wsrcL[row * D_DIM + cw];
  }
#endif

  // ---- stage x tile into LDS as bf16 hi/lo ----
#pragma unroll 4
  for (int row = 0; row < BT; ++row) {
    float v = x[((size_t)(b0 + row) * N_GRP + n) * D_DIM + t];
    ushort_t hb, lb;
    split_bits(v, hb, lb);
    sXh[row * WPAD + t] = hb;
    sXl[row * WPAD + t] = lb;
  }

  // ---- fp32 residual registers in C/D-matrix layout + bias ----
  v8f cur[2][2];
  float bias[2];
#pragma unroll
  for (int jt = 0; jt < 2; ++jt) {
    int j = jr * 32 + jt * 16 + ln15;
    bias[jt] = gB[n * D_DIM + j];
#pragma unroll
    for (int mt = 0; mt < 2; ++mt)
#pragma unroll
      for (int r = 0; r < 8; ++r) {
        int m = mt * 16 + lhi * 8 + r;
        cur[mt][jt][r] = x[((size_t)(b0 + m) * N_GRP + n) * D_DIM + j];
      }
  }

#if HAVE_ASYNC_LDS
  __builtin_amdgcn_s_wait_asynccnt(0);
#endif
  __syncthreads();

  for (int step = 0; step < 4; ++step) {
    v8f acc[2][2];
#pragma unroll
    for (int mt = 0; mt < 2; ++mt)
#pragma unroll
      for (int jt = 0; jt < 2; ++jt)
#pragma unroll
        for (int r = 0; r < 8; ++r) acc[mt][jt][r] = 0.0f;

#pragma unroll
    for (int kc = 0; kc < 8; ++kc) {
      // A fragments (16x32 bf16) for both m-tiles
      v16bf Ah[2], Al[2];
#pragma unroll
      for (int mt = 0; mt < 2; ++mt) {
        int abase = (mt * 16 + ln15) * WPAD + kc * 32 + lhi * 16;
        v8bf a0h = *(const v8bf*)&sXh[abase];
        v8bf a1h = *(const v8bf*)&sXh[abase + 8];
        v8bf a0l = *(const v8bf*)&sXl[abase];
        v8bf a1l = *(const v8bf*)&sXl[abase + 8];
        Ah[mt] = __builtin_shufflevector(a0h, a1h, 0,1,2,3,4,5,6,7,8,9,10,11,12,13,14,15);
        Al[mt] = __builtin_shufflevector(a0l, a1l, 0,1,2,3,4,5,6,7,8,9,10,11,12,13,14,15);
      }
      // B fragments (32x16 bf16) for both j-tiles
      v16bf Bh[2], Bl[2];
#pragma unroll
      for (int jt = 0; jt < 2; ++jt) {
        int bbase = (jr * 32 + jt * 16 + ln15) * WPAD + kc * 32 + lhi * 16;
        Bh[jt] = *(const v16bf*)&sWh[bbase];
        Bl[jt] = *(const v16bf*)&sWl[bbase];
      }
      // term-major issue order: 4 independent accumulation chains per term
#pragma unroll
      for (int mt = 0; mt < 2; ++mt)
#pragma unroll
        for (int jt = 0; jt < 2; ++jt)
          acc[mt][jt] = wmma_bf16(Ah[mt], Bh[jt], acc[mt][jt]);   // hi*hi
#pragma unroll
      for (int mt = 0; mt < 2; ++mt)
#pragma unroll
        for (int jt = 0; jt < 2; ++jt)
          acc[mt][jt] = wmma_bf16(Ah[mt], Bl[jt], acc[mt][jt]);   // hi*lo
#pragma unroll
      for (int mt = 0; mt < 2; ++mt)
#pragma unroll
        for (int jt = 0; jt < 2; ++jt)
          acc[mt][jt] = wmma_bf16(Al[mt], Bh[jt], acc[mt][jt]);   // lo*hi
    }

    __syncthreads();   // all x reads for this step complete

    // residual + relu in exact fp32
#pragma unroll
    for (int mt = 0; mt < 2; ++mt)
#pragma unroll
      for (int jt = 0; jt < 2; ++jt)
#pragma unroll
        for (int r = 0; r < 8; ++r) {
          float z = acc[mt][jt][r] + bias[jt];
          cur[mt][jt][r] += (z > 0.0f ? z : 0.0f);
        }

    if (step < 3) {
      // write updated tile back to LDS as bf16 hi/lo
#pragma unroll
      for (int mt = 0; mt < 2; ++mt)
#pragma unroll
        for (int jt = 0; jt < 2; ++jt) {
          int j = jr * 32 + jt * 16 + ln15;
#pragma unroll
          for (int r = 0; r < 8; ++r) {
            int m = mt * 16 + lhi * 8 + r;
            ushort_t hb, lb;
            split_bits(cur[mt][jt][r], hb, lb);
            sXh[m * WPAD + j] = hb;
            sXl[m * WPAD + j] = lb;
          }
        }
    }
    __syncthreads();
  }

  // final fp32 store
#pragma unroll
  for (int mt = 0; mt < 2; ++mt)
#pragma unroll
    for (int jt = 0; jt < 2; ++jt) {
      int j = jr * 32 + jt * 16 + ln15;
#pragma unroll
      for (int r = 0; r < 8; ++r) {
        int m = mt * 16 + lhi * 8 + r;
        out[((size_t)(b0 + m) * N_GRP + n) * D_DIM + j] = cur[mt][jt][r];
      }
    }
}

extern "C" void kernel_launch(void* const* d_in, const int* in_sizes, int n_in,
                              void* d_out, int out_size, void* d_ws, size_t ws_size,
                              hipStream_t stream) {
  const float* x     = (const float*)d_in[0];
  const float* w     = (const float*)d_in[1];
  const float* b     = (const float*)d_in[2];
  const float* wmask = (const float*)d_in[3];
  const float* bmask = (const float*)d_in[4];
  float* out = (float*)d_out;

  const int NW = N_GRP * D_DIM * D_DIM;   // 4,194,304
  const int NB = N_GRP * D_DIM;           // 16,384

  // ws layout: [wh bf16 bits NW][wl bf16 bits NW][bias f32 NB]  (~16.1 MB)
  ushort_t* wh = (ushort_t*)d_ws;
  ushort_t* wl = wh + NW;
  float* bb = (float*)(wl + NW);

  prep_w_kernel<<<(NW + TPB - 1) / TPB, TPB, 0, stream>>>(w, wmask, wh, wl, NW);
  prep_b_kernel<<<(NB + TPB - 1) / TPB, TPB, 0, stream>>>(b, bmask, bb, NB);

  dim3 grid(B_TOT / BT, N_GRP);
  BatchLinearMasked_recur_kernel<<<grid, TPB, 0, stream>>>(x, wh, wl, bb, out);
}
